// Meta_Transformer_Layer_21208548508195
// MI455X (gfx1250) — matrix operand order repaired and checked by reading, not compiled, not run on hardware
//
#include <hip/hip_runtime.h>
#include <hip/hip_bf16.h>

// ---------------------------------------------------------------------------
// Types for CDNA5 WMMA (wave32)
// ---------------------------------------------------------------------------
typedef __bf16 bf16_t;
typedef __attribute__((ext_vector_type(16))) __bf16 v16bf;
typedef __attribute__((ext_vector_type(8)))  __bf16 v8bf;
typedef __attribute__((ext_vector_type(8)))  float  v8f;
typedef __attribute__((ext_vector_type(4)))  unsigned int v4u;
typedef __attribute__((ext_vector_type(8)))  int v8i;
typedef __attribute__((ext_vector_type(4)))  int v4i;

#define E_DIM 1024
#define B_DIM 32
#define F_DIM 512
#define H_DIM 16
#define D_HEAD 64
#define M_DIM (B_DIM * F_DIM)   // 16384 rows

// Tensor Data Mover availability (device pass only; 6-arg clang-23 form)
#if defined(__AMDGCN__) && __has_builtin(__builtin_amdgcn_tensor_load_to_lds) && \
    __has_builtin(__builtin_amdgcn_s_wait_tensorcnt)
#define USE_TDM 1
#else
#define USE_TDM 0
#endif

__device__ __forceinline__ v8f wmma_bf16(v16bf a, v16bf b, v8f c) {
  // D = A(16x32 bf16) * B(32x16 bf16) + C(16x16 f32)
  return __builtin_amdgcn_wmma_f32_16x16x32_bf16(
      /*neg_a=*/false, a, /*neg_b=*/false, b,
      /*c_mod=*/(short)0, c, /*reuse_a=*/false, /*reuse_b=*/false);
}

__device__ __forceinline__ v16bf cat8(v8bf lo, v8bf hi) {
  return __builtin_shufflevector(lo, hi, 0,1,2,3,4,5,6,7,8,9,10,11,12,13,14,15);
}

// Flat LDS pointer -> LDS byte offset (ISA 10.2: low 32 bits of the address)
__device__ __forceinline__ unsigned ldsoff(const void* p) {
  return (unsigned)(uintptr_t)p;
}

#if USE_TDM
// ---------------------------------------------------------------------------
// TDM 2D tile load: tile (tile_w x tile_h) of bf16, row stride row_stride
// elements, from gsrc (tile start) into LDS at lds_byte_off.
// D# packing per CDNA5 ISA.
// 6-arg builtin: (uint32x4 g0, int32x8 g1, int32x4, int32x4, int32x8, i32 cpol)
// ---------------------------------------------------------------------------
__device__ __forceinline__ void tdm_load_tile_2d(unsigned lds_byte_off,
                                                 const bf16_t* gsrc,
                                                 unsigned tile_w, unsigned tile_h,
                                                 unsigned row_stride,
                                                 unsigned tensor_w, unsigned tensor_h) {
  unsigned long long ga = (unsigned long long)(uintptr_t)gsrc;
  v4u g0;
  g0[0] = 1u;                                        // count=1, user descriptor
  g0[1] = lds_byte_off;                              // lds_addr
  g0[2] = (unsigned)ga;                              // global_addr[31:0]
  g0[3] = (unsigned)((ga >> 32) & 0x01ffffffu)       // global_addr[56:32]
        | (2u << 30);                                // type = 2 ("image")
  v8i g1;
  g1[0] = (int)(1u << 16);                           // data_size=1 (2B), no mcast
  g1[1] = (int)((tensor_w & 0xffffu) << 16);         // tensor_dim0[15:0] @ bit48
  g1[2] = (int)((tensor_w >> 16) | ((tensor_h & 0xffffu) << 16));  // dim0 hi | dim1 lo
  g1[3] = (int)((tensor_h >> 16) | ((tile_w & 0xffffu) << 16));    // dim1 hi | tile0
  g1[4] = (int)(tile_h & 0xffffu);                   // tile_dim1 (tile_dim2=0)
  g1[5] = (int)row_stride;                           // tensor_dim0_stride[31:0]
  g1[6] = 0;                                         // stride[47:32]=0, dim1_stride lo
  g1[7] = 0;
  v4i g2 = {0, 0, 0, 0};                             // 2D: dims 2/3 unused
  v4i g3 = {0, 0, 0, 0};
  v8i g4 = {0, 0, 0, 0, 0, 0, 0, 0};                 // extra group (zero)
  __builtin_amdgcn_tensor_load_to_lds(g0, g1, g2, g3, g4, 0);
}
#endif

// ---------------------------------------------------------------------------
// fp32 -> bf16 convert (grid-stride)
// ---------------------------------------------------------------------------
__global__ void k_cvt(const float* __restrict__ in, bf16_t* __restrict__ out, size_t n) {
  size_t i = (size_t)blockIdx.x * blockDim.x + threadIdx.x;
  size_t stride = (size_t)gridDim.x * blockDim.x;
  for (; i < n; i += stride) out[i] = (bf16_t)in[i];
}

// fp32 [R][C] -> bf16 transposed [C][R]  (for W_out: out = x @ W_out^T)
__global__ void k_cvt_t(const float* __restrict__ in, bf16_t* __restrict__ out, int R, int C) {
  size_t n = (size_t)R * C;
  size_t i = (size_t)blockIdx.x * blockDim.x + threadIdx.x;
  size_t stride = (size_t)gridDim.x * blockDim.x;
  for (; i < n; i += stride) {
    int r = (int)(i / C);
    int c = (int)(i % C);
    out[(size_t)c * R + r] = (bf16_t)in[i];
  }
}

// ---------------------------------------------------------------------------
// WMMA bf16 GEMM: C[M,N] = A[M,K] * B[K,N]  (+resid, +relu)
// Double-buffered LDS; A tiles via TDM (wave 0) when available; B tiles
// transposed to [n][k] so fragments are contiguous 32B LDS reads.
// Optional second bf16 output C2 (fused convert).
// Block tile 128x128, BK=32, 256 threads (8 waves), wave tile 64x32.
// ---------------------------------------------------------------------------
#define BM 128
#define BN 128
#define BK 32

template <bool RELU, typename OutT>
__global__ __launch_bounds__(256) void k_gemm(const bf16_t* __restrict__ A,
                                              const bf16_t* __restrict__ Bm,
                                              OutT* __restrict__ C,
                                              bf16_t* __restrict__ C2,
                                              const float* __restrict__ resid,
                                              int M, int N, int K) {
  __shared__ bf16_t As[2][BM][BK];      // [buf][m][k]
  __shared__ bf16_t Bs[2][BN][BK];      // [buf][n][k] (transposed tile)

  const int tid  = threadIdx.x;
  const int wave = tid >> 5;
  const int lane = tid & 31;
  const int hlf  = lane >> 4;           // 0: lanes 0-15, 1: lanes 16-31
  const int l16  = lane & 15;

  const int nblk = N / BN;
  const int m0 = (blockIdx.x / nblk) * BM;
  const int n0 = (blockIdx.x % nblk) * BN;
  const int wm = (wave & 1) * 64;       // 2 wave rows of 64
  const int wn = (wave >> 1) * 32;      // 4 wave cols of 32

  v8f acc[4][2];
#pragma unroll
  for (int i = 0; i < 4; i++)
#pragma unroll
    for (int j = 0; j < 2; j++) acc[i][j] = (v8f){};

  // cooperative-load coordinates
  const int ar = tid >> 1;              // 0..127
  const int ac = (tid & 1) * 16;        // 0 or 16
  const int bk = tid >> 3;              // 0..31
  const int bn = (tid & 7) * 16;        // 0..112

  // ---- prologue: stage tile 0 into buffer 0 ----
  {
#if USE_TDM
    if (wave == 0)
      tdm_load_tile_2d(ldsoff(&As[0][0][0]), A + (size_t)m0 * K,
                       BK, BM, (unsigned)K, (unsigned)K, (unsigned)M);
#else
    const bf16_t* srcA = A + (size_t)(m0 + ar) * K + ac;
    *(v8bf*)&As[0][ar][ac]     = *(const v8bf*)srcA;
    *(v8bf*)&As[0][ar][ac + 8] = *(const v8bf*)(srcA + 8);
#endif
    const bf16_t* srcB = Bm + (size_t)bk * N + n0 + bn;
    v8bf x0 = *(const v8bf*)srcB;
    v8bf x1 = *(const v8bf*)(srcB + 8);
#pragma unroll
    for (int e = 0; e < 8; e++) Bs[0][bn + e][bk]     = x0[e];
#pragma unroll
    for (int e = 0; e < 8; e++) Bs[0][bn + 8 + e][bk] = x1[e];
#if USE_TDM
    if (wave == 0) __builtin_amdgcn_s_wait_tensorcnt(0);
#endif
  }
  __syncthreads();

  int cur = 0;
  for (int k0 = 0; k0 < K; k0 += BK) {
    const bool more = (k0 + BK) < K;
    const int nxt = cur ^ 1;
    v8bf nb0 = {}, nb1 = {};
#if !USE_TDM
    v8bf na0 = {}, na1 = {};
#endif
    if (more) {
      // issue next tile's loads before compute (overlap with WMMA pipe)
#if USE_TDM
      if (wave == 0)
        tdm_load_tile_2d(ldsoff(&As[nxt][0][0]), A + (size_t)m0 * K + (k0 + BK),
                         BK, BM, (unsigned)K, (unsigned)K, (unsigned)M);
#else
      const bf16_t* srcA = A + (size_t)(m0 + ar) * K + (k0 + BK) + ac;
      na0 = *(const v8bf*)srcA;
      na1 = *(const v8bf*)(srcA + 8);
#endif
      const bf16_t* srcB = Bm + (size_t)(k0 + BK + bk) * N + n0 + bn;
      nb0 = *(const v8bf*)srcB;
      nb1 = *(const v8bf*)(srcB + 8);
    }

    // ---- compute on buffer `cur` ----
    // B fragments: element e <-> B[k = 16*hlf + e][col]
    v16bf bfr[2];
#pragma unroll
    for (int tn = 0; tn < 2; tn++) {
      const int col = wn + tn * 16 + l16;
      bfr[tn] = *(const v16bf*)&Bs[cur][col][hlf * 16];
    }
#pragma unroll
    for (int tm = 0; tm < 4; tm++) {
      const int row = wm + tm * 16 + l16;
      v16bf af = *(const v16bf*)&As[cur][row][hlf * 16];
      acc[tm][0] = wmma_bf16(af, bfr[0], acc[tm][0]);
      acc[tm][1] = wmma_bf16(af, bfr[1], acc[tm][1]);
    }

    if (more) {
      // stash next B tile (transposed) into the other buffer
#pragma unroll
      for (int e = 0; e < 8; e++) Bs[nxt][bn + e][bk]     = nb0[e];
#pragma unroll
      for (int e = 0; e < 8; e++) Bs[nxt][bn + 8 + e][bk] = nb1[e];
#if !USE_TDM
      *(v8bf*)&As[nxt][ar][ac]     = na0;
      *(v8bf*)&As[nxt][ar][ac + 8] = na1;
#else
      if (wave == 0) __builtin_amdgcn_s_wait_tensorcnt(0);
#endif
    }
    __syncthreads();
    cur = nxt;
  }

  // Epilogue: C layout row = v + 8*hlf, col = l16 within each 16x16 tile
#pragma unroll
  for (int tm = 0; tm < 4; tm++) {
#pragma unroll
    for (int tn = 0; tn < 2; tn++) {
      const int col = n0 + wn + tn * 16 + l16;
#pragma unroll
      for (int v = 0; v < 8; v++) {
        const int row = m0 + wm + tm * 16 + v + hlf * 8;
        float val = acc[tm][tn][v];
        if (resid) val += resid[(size_t)row * N + col];
        if (RELU) val = fmaxf(val, 0.0f);
        C[(size_t)row * N + col] = (OutT)val;
        if (C2) C2[(size_t)row * N + col] = (bf16_t)val;
      }
    }
  }
}

// ---------------------------------------------------------------------------
// LayerNorm over last dim N=1024, one block (256 thr) per row.
// ---------------------------------------------------------------------------
template <typename OutT>
__global__ __launch_bounds__(256) void k_layernorm(const float* __restrict__ x,
                                                   const float* __restrict__ g,
                                                   const float* __restrict__ bb,
                                                   OutT* __restrict__ y, int N) {
  __shared__ float redS[8], redQ[8];
  const int row = blockIdx.x;
  const float* xr = x + (size_t)row * N;
  const int t4 = threadIdx.x * 4;

  float4 v = *(const float4*)(xr + t4);
  float s = v.x + v.y + v.z + v.w;
  float q = v.x * v.x + v.y * v.y + v.z * v.z + v.w * v.w;
#pragma unroll
  for (int off = 16; off > 0; off >>= 1) {
    s += __shfl_xor(s, off, 32);
    q += __shfl_xor(q, off, 32);
  }
  const int wave = threadIdx.x >> 5, lane = threadIdx.x & 31;
  if (lane == 0) { redS[wave] = s; redQ[wave] = q; }
  __syncthreads();
  float ts = 0.f, tq = 0.f;
#pragma unroll
  for (int w = 0; w < 8; w++) { ts += redS[w]; tq += redQ[w]; }
  const float mu  = ts / N;
  const float var = tq / N - mu * mu;
  const float r   = rsqrtf(var + 1e-6f);
  float4 gw = *(const float4*)(g + t4);
  float4 bw = *(const float4*)(bb + t4);
  OutT* yr = y + (size_t)row * N + t4;
  yr[0] = (OutT)((v.x - mu) * r * gw.x + bw.x);
  yr[1] = (OutT)((v.y - mu) * r * gw.y + bw.y);
  yr[2] = (OutT)((v.z - mu) * r * gw.z + bw.z);
  yr[3] = (OutT)((v.w - mu) * r * gw.w + bw.w);
}

// ---------------------------------------------------------------------------
// Flash attention, WMMA, online softmax.
// Grid: B*H*(F/128) blocks, 256 threads (8 waves). Wave owns 16 Q rows.
// Q/K/V/O are bf16 [B*F, E], head h occupies columns [h*64, h*64+64).
// ---------------------------------------------------------------------------
__global__ __launch_bounds__(256) void k_attn(const bf16_t* __restrict__ Q,
                                              const bf16_t* __restrict__ K,
                                              const bf16_t* __restrict__ V,
                                              bf16_t* __restrict__ O) {
  __shared__ bf16_t Ks[32][D_HEAD];        // [j_local][d]
  __shared__ bf16_t Vst[D_HEAD][32];       // transposed: [d][j_local]
  __shared__ bf16_t Ps[8][16][32];         // per-wave P tile (C->A layout relay)

  const int tid  = threadIdx.x;
  const int wave = tid >> 5;
  const int lane = tid & 31;
  const int hlf  = lane >> 4;
  const int l16  = lane & 15;

  const int qblk = blockIdx.x % (F_DIM / 128);
  const int h    = (blockIdx.x / (F_DIM / 128)) % H_DIM;
  const int b    = blockIdx.x / ((F_DIM / 128) * H_DIM);

  const size_t baseRow = (size_t)b * F_DIM;
  const int hcol = h * D_HEAD;
  const int q0   = qblk * 128 + wave * 16;

  // Q fragments (A layout), frag f covers k in [32f, 32f+32)
  v16bf qf[2];
#pragma unroll
  for (int f = 0; f < 2; f++) {
    const bf16_t* src = Q + (baseRow + q0 + l16) * E_DIM + hcol + 32 * f;
    qf[f] = *(const v16bf*)(src + hlf * 16);
  }

  v8f o[4];
#pragma unroll
  for (int c = 0; c < 4; c++) o[c] = (v8f){};
  float mrow[8], lrow[8];
#pragma unroll
  for (int v = 0; v < 8; v++) { mrow[v] = -1e30f; lrow[v] = 0.0f; }

  const float scale = 0.125f;  // 1/sqrt(64)

  const int cr = tid >> 3;           // 0..31  (key row)
  const int cc = (tid & 7) * 8;      // 0..56  (d col)

  for (int j0 = 0; j0 < F_DIM; j0 += 32) {
    __syncthreads();   // protect LDS tiles from previous iteration's readers
    {  // K tile straight, V tile transposed
      v8bf kx = *(const v8bf*)(K + (baseRow + j0 + cr) * E_DIM + hcol + cc);
      *(v8bf*)&Ks[cr][cc] = kx;
      v8bf vx = *(const v8bf*)(V + (baseRow + j0 + cr) * E_DIM + hcol + cc);
#pragma unroll
      for (int e = 0; e < 8; e++) Vst[cc + e][cr] = vx[e];
    }
    __syncthreads();

    // S = Q * K^T  (two 16x16 column tiles, each accumulates 2 WMMAs over d)
    v8f s[2];
#pragma unroll
    for (int t = 0; t < 2; t++) {
      v16bf k0f = *(const v16bf*)&Ks[16 * t + l16][hlf * 16];        // k in [0,32)
      v16bf k1f = *(const v16bf*)&Ks[16 * t + l16][32 + hlf * 16];   // k in [32,64)
      s[t] = wmma_bf16(qf[0], k0f, (v8f){});
      s[t] = wmma_bf16(qf[1], k1f, s[t]);
#pragma unroll
      for (int v = 0; v < 8; v++) s[t][v] *= scale;
    }

    // online softmax per row (row = v + 8*hlf; 16 cols live in 16-lane half)
#pragma unroll
    for (int v = 0; v < 8; v++) {
      float mx = fmaxf(s[0][v], s[1][v]);
#pragma unroll
      for (int off = 1; off < 16; off <<= 1) mx = fmaxf(mx, __shfl_xor(mx, off, 32));
      const float mnew  = fmaxf(mrow[v], mx);
      const float alpha = __expf(mrow[v] - mnew);
      mrow[v] = mnew;
      const float p0 = __expf(s[0][v] - mnew);
      const float p1 = __expf(s[1][v] - mnew);
      s[0][v] = p0; s[1][v] = p1;
      float rs = p0 + p1;
#pragma unroll
      for (int off = 1; off < 16; off <<= 1) rs += __shfl_xor(rs, off, 32);
      lrow[v] = lrow[v] * alpha + rs;
#pragma unroll
      for (int c = 0; c < 4; c++) o[c][v] *= alpha;
    }

    // P: C layout -> LDS -> A layout (per-wave private region, DS in-order)
#pragma unroll
    for (int v = 0; v < 8; v++) {
      Ps[wave][v + 8 * hlf][l16]      = (bf16_t)s[0][v];
      Ps[wave][v + 8 * hlf][16 + l16] = (bf16_t)s[1][v];
    }
    v16bf pfrag = *(const v16bf*)&Ps[wave][l16][hlf * 16];

    // O += P * V   (4 column tiles of d)
#pragma unroll
    for (int c = 0; c < 4; c++) {
      v16bf vf = *(const v16bf*)&Vst[16 * c + l16][hlf * 16];
      o[c] = wmma_bf16(pfrag, vf, o[c]);
    }
  }

  // finalize and store
#pragma unroll
  for (int c = 0; c < 4; c++) {
#pragma unroll
    for (int v = 0; v < 8; v++) {
      const float val = o[c][v] / lrow[v];
      const size_t row = baseRow + q0 + v + 8 * hlf;
      O[row * E_DIM + hcol + 16 * c + l16] = (bf16_t)val;
    }
  }
}

// ---------------------------------------------------------------------------
// Host orchestration
// ---------------------------------------------------------------------------
extern "C" void kernel_launch(void* const* d_in, const int* in_sizes, int n_in,
                              void* d_out, int out_size, void* d_ws, size_t ws_size,
                              hipStream_t stream) {
  (void)in_sizes; (void)n_in; (void)out_size; (void)ws_size;

  const float* x_in     = (const float*)d_in[0];   // [B,F,E]
  const float* W_Q      = (const float*)d_in[1];
  const float* W_K      = (const float*)d_in[2];
  const float* W_V      = (const float*)d_in[3];
  const float* W_out    = (const float*)d_in[4];
  const float* ln_w     = (const float*)d_in[5];
  const float* ln_b     = (const float*)d_in[6];
  const float* mlp_w0   = (const float*)d_in[7];
  const float* mlp_w1   = (const float*)d_in[8];
  const float* ffn_ln_w = (const float*)d_in[9];
  const float* ffn_ln_b = (const float*)d_in[10];
  float* out = (float*)d_out;

  const int M = M_DIM, E = E_DIM;
  const size_t ME = (size_t)M * E;
  const size_t EE = (size_t)E * E;

  char* ws = (char*)d_ws;
  size_t off = 0;
  auto alloc = [&](size_t bytes) -> void* {
    void* p = ws + off;
    off += (bytes + 255) & ~(size_t)255;
    return p;
  };

  bf16_t* Xb   = (bf16_t*)alloc(ME * 2);
  bf16_t* Wqb  = (bf16_t*)alloc(EE * 2);
  bf16_t* Wkb  = (bf16_t*)alloc(EE * 2);
  bf16_t* Wvb  = (bf16_t*)alloc(EE * 2);
  bf16_t* W0b  = (bf16_t*)alloc(EE * 2);
  bf16_t* W1b  = (bf16_t*)alloc(EE * 2);
  bf16_t* WoTb = (bf16_t*)alloc(EE * 2);
  float*  Qf   = (float*) alloc(ME * 4);   // reused later for Of
  bf16_t* Qb   = (bf16_t*)alloc(ME * 2);
  bf16_t* Kb   = (bf16_t*)alloc(ME * 2);
  bf16_t* Vb   = (bf16_t*)alloc(ME * 2);
  bf16_t* Hb   = (bf16_t*)alloc(ME * 2);
  float*  Q2f  = (float*) alloc(ME * 4);
  bf16_t* Qlnb = (bf16_t*)alloc(ME * 2);
  bf16_t* Ab   = (bf16_t*)alloc(ME * 2);
  float*  Of   = Qf;                        // Qf free after Q2f GEMM

  const dim3 blk(256);
  const dim3 gGemm((M / BM) * (E / BN));    // 1024 blocks
  const dim3 gAttn(B_DIM * H_DIM * (F_DIM / 128));
  const dim3 gLN(M);

  // 1) bf16 conversions
  k_cvt<<<2048, blk, 0, stream>>>(x_in, Xb, ME);
  k_cvt<<<256,  blk, 0, stream>>>(W_Q,    Wqb, EE);
  k_cvt<<<256,  blk, 0, stream>>>(W_K,    Wkb, EE);
  k_cvt<<<256,  blk, 0, stream>>>(W_V,    Wvb, EE);
  k_cvt<<<256,  blk, 0, stream>>>(mlp_w0, W0b, EE);
  k_cvt<<<256,  blk, 0, stream>>>(mlp_w1, W1b, EE);
  k_cvt_t<<<256, blk, 0, stream>>>(W_out, WoTb, E, E);

  // 2) QKV projections (Q also emits fused bf16 copy)
  k_gemm<false, float ><<<gGemm, blk, 0, stream>>>(Xb, Wqb, Qf, Qb,      nullptr, M, E, E);
  k_gemm<false, bf16_t><<<gGemm, blk, 0, stream>>>(Xb, Wkb, Kb, nullptr, nullptr, M, E, E);
  k_gemm<false, bf16_t><<<gGemm, blk, 0, stream>>>(Xb, Wvb, Vb, nullptr, nullptr, M, E, E);

  // 3) meta FFN on Q: q + relu(q@w0)@w1, then LayerNorm(ffn_ln)
  k_gemm<true,  bf16_t><<<gGemm, blk, 0, stream>>>(Qb, W0b, Hb,  nullptr, nullptr, M, E, E);
  k_gemm<false, float ><<<gGemm, blk, 0, stream>>>(Hb, W1b, Q2f, nullptr, Qf,      M, E, E);
  k_layernorm<bf16_t><<<gLN, blk, 0, stream>>>(Q2f, ffn_ln_w, ffn_ln_b, Qlnb, E);

  // 4) attention
  k_attn<<<gAttn, blk, 0, stream>>>(Qlnb, Kb, Vb, Ab);

  // 5) output projection + residual + LayerNorm
  k_gemm<false, float><<<gGemm, blk, 0, stream>>>(Ab, WoTb, Of, nullptr, x_in, M, E, E);
  k_layernorm<float><<<gLN, blk, 0, stream>>>(Of, ln_w, ln_b, out, E);
}
